// THPEncoder_73212012527972
// MI455X (gfx1250) — matrix operand order repaired and checked by reading, not compiled, not run on hardware
//
#include <hip/hip_runtime.h>

typedef __attribute__((ext_vector_type(16))) _Float16 v16h;
typedef __attribute__((ext_vector_type(8)))  _Float16 v8h;
typedef __attribute__((ext_vector_type(8)))  float    v8f;

#define Bc 16
#define Sc 1024
#define Dc 512
#define Hc 4

// Pin an interleaved WMMA/VMEM schedule (depth-1 software pipeline) if the
// scheduling intrinsics exist; otherwise no-op.  Mask: 0x008 = MFMA/WMMA,
// 0x020 = VMEM read.
#if defined(__has_builtin)
#if __has_builtin(__builtin_amdgcn_sched_group_barrier)
#define PIPELINE_HINT()                                   \
    do {                                                  \
        __builtin_amdgcn_sched_group_barrier(0x008, 1, 0);\
        __builtin_amdgcn_sched_group_barrier(0x020, 3, 0);\
        __builtin_amdgcn_sched_group_barrier(0x008, 1, 0);\
        __builtin_amdgcn_sched_group_barrier(0x020, 3, 0);\
        __builtin_amdgcn_sched_group_barrier(0x008, 1, 0);\
        __builtin_amdgcn_sched_group_barrier(0x020, 2, 0);\
        __builtin_amdgcn_sched_group_barrier(0x008, 1, 0);\
        __builtin_amdgcn_sched_group_barrier(0x020, 2, 0);\
    } while (0)
#endif
#endif
#ifndef PIPELINE_HINT
#define PIPELINE_HINT() do { } while (0)
#endif

__device__ __forceinline__ v8f wmma_f16(v16h a, v16h b, v8f c) {
    // (neg_a, A, neg_b, B, c_mod, C, reuse_a, reuse_b)
    return __builtin_amdgcn_wmma_f32_16x16x32_f16(false, a, false, b, (short)0, c,
                                                  false, false);
}

__device__ __forceinline__ v16h make_a(const _Float16* p) {
    v8h a0 = *(const v8h*)(p);
    v8h a1 = *(const v8h*)(p + 8);
    v16h a;
#pragma unroll
    for (int j = 0; j < 8; j++) { a[j] = a0[j]; a[j + 8] = a1[j]; }
    return a;
}

// ---------------------------------------------------------------------------
// Prep: zero output accumulator, convert x -> f16, convert W -> f16
// ---------------------------------------------------------------------------
__global__ void prep_kernel(const float* __restrict__ xin,
                            const float* __restrict__ Wf,
                            float* __restrict__ out,
                            _Float16* __restrict__ xh,
                            _Float16* __restrict__ Wh) {
    long i = (long)blockIdx.x * blockDim.x + threadIdx.x;
    const long NX = (long)Bc * Sc * Dc;       // 8,388,608
    const long NW = (long)Hc * Dc * Dc;       // 1,048,576
    if (i < NX) { out[i] = 0.0f; xh[i] = (_Float16)xin[i]; }
    if (i < NW) { Wh[i] = (_Float16)Wf[i]; }
}

// ---------------------------------------------------------------------------
// Value path: vT[b][e][s] = f16( ELU( sum_d xh[b][s][d] * Wh[e][d] + bias[e] ) )
// Each wave computes 16(M=s) x 64(N=e); B fragments hoisted before WMMA group.
// ---------------------------------------------------------------------------
__global__ void __launch_bounds__(256, 4)
value_kernel(const _Float16* __restrict__ xh,
             const _Float16* __restrict__ Wh,   // head slice [D][D]
             const float*    __restrict__ bias, // head slice [D]
             _Float16* __restrict__ vT) {
    const int lane = threadIdx.x & 31;
    const int wave = threadIdx.x >> 5;
    const int wid  = blockIdx.x * 8 + wave;        // 0 .. 8191
    const int b    = wid >> 9;                     // / (64*8)
    const int rr_  = wid & 511;
    const int m0   = (rr_ >> 3) * 16;              // s tile
    const int n0   = (rr_ & 7) * 64;               // e strip (4 x 16)
    const int hi   = lane >> 4, lo = lane & 15;

    const _Float16* Arow = xh + (size_t)(b * Sc + m0 + lo) * Dc;
    const _Float16* Brow = Wh + (size_t)(n0 + lo) * Dc;   // +j*16*Dc per sub-tile

    v8f acc[4] = {};
#pragma unroll 4
    for (int kb = 0; kb < Dc; kb += 32) {
        v16h a = make_a(Arow + kb + hi * 16);
        v16h bm[4];
#pragma unroll
        for (int j = 0; j < 4; j++)
            bm[j] = *(const v16h*)(Brow + (size_t)j * 16 * Dc + kb + hi * 16);
#pragma unroll
        for (int j = 0; j < 4; j++)
            acc[j] = wmma_f16(a, bm[j], acc[j]);
        PIPELINE_HINT();
    }

#pragma unroll
    for (int j = 0; j < 4; j++) {
        const int   e  = n0 + j * 16 + lo;
        const float bv = bias[e];
        v8h st;
#pragma unroll
        for (int r = 0; r < 8; r++) {
            float z = acc[j][r] + bv;
            z = (z > 0.0f) ? z : expm1f(z);        // ELU, alpha = 1
            st[r] = (_Float16)z;
        }
        // D-layout: lane holds col N, rows m = r + hi*8 (consecutive s) -> 16B store
        *(v8h*)(vT + (size_t)(b * Dc + e) * Sc + m0 + hi * 8) = st;
    }
}

// ---------------------------------------------------------------------------
// Attention: per (b, 16-query tile) workgroup.
//   Phase 1: scores[16][1024] = mask * (x_q . x_k) / sqrt(D)  (WMMA, f16 LDS)
//            2 passes of 4 key tiles per wave (4 live accumulators each).
//   Phase 2: row L2 norms -> invn[16]
//   Phase 4: x_new[q] = invn[q] * (attn_unnorm @ v)  (norm folded into epilogue)
// ---------------------------------------------------------------------------
__global__ void __launch_bounds__(256, 4)
attn_kernel(const _Float16* __restrict__ xh,
            const _Float16* __restrict__ vT,
            const int*      __restrict__ evt,
            float*          __restrict__ out,
            _Float16*       __restrict__ xh_next) {
    __shared__ __align__(32) _Float16 attnh[16][Sc + 8];
    __shared__ float red[16][16];
    __shared__ float invn[16];

    const int lane = threadIdx.x & 31;
    const int wave = threadIdx.x >> 5;
    const int b    = blockIdx.x >> 6;
    const int q0   = (blockIdx.x & 63) * 16;
    const int hi   = lane >> 4, lo = lane & 15;
    const float invT = 0.044194173824159216f;      // 1/sqrt(512)

    const _Float16* Arow = xh + (size_t)(b * Sc + q0 + lo) * Dc;

    // ---- Phase 1: masked, scaled scores into LDS (f16) ----
    for (int p = 0; p < 2; p++) {                  // 2 passes x 4 key tiles
        const int kbase = wave * 128 + p * 64;
        const _Float16* Brow = xh + (size_t)(b * Sc + kbase + lo) * Dc; // +j*16*Dc

        v8f acc[4] = {};
#pragma unroll 4
        for (int kb = 0; kb < Dc; kb += 32) {
            v16h a = make_a(Arow + kb + hi * 16);
            v16h bm[4];
#pragma unroll
            for (int j = 0; j < 4; j++)
                bm[j] = *(const v16h*)(Brow + (size_t)j * 16 * Dc + kb + hi * 16);
#pragma unroll
            for (int j = 0; j < 4; j++)
                acc[j] = wmma_f16(a, bm[j], acc[j]);
            PIPELINE_HINT();
        }

#pragma unroll
        for (int j = 0; j < 4; j++) {
            const int k   = kbase + j * 16 + lo;        // key index for this lane
            const int pad = (evt[b * Sc + k] == 0);     // PAD == 0
#pragma unroll
            for (int r = 0; r < 8; r++) {
                const int m = r + hi * 8;               // query row within tile
                const int q = q0 + m;
                // reference keeps entries where mask==1: future (k>q) or pad key
                float val = ((k > q) || pad) ? acc[j][r] * invT : 0.0f;
                attnh[m][k] = (_Float16)val;
            }
        }
    }
    __syncthreads();

    // ---- Phase 2: row L2 norms ----
    {
        const int row = threadIdx.x >> 4;
        const int c   = threadIdx.x & 15;
        float s = 0.0f;
        for (int j = c; j < Sc; j += 16) {
            float v = (float)attnh[row][j];
            s += v * v;
        }
        red[row][c] = s;
    }
    __syncthreads();
    if (threadIdx.x < 16) {
        float s = 0.0f;
#pragma unroll
        for (int j = 0; j < 16; j++) s += red[threadIdx.x][j];
        invn[threadIdx.x] = 1.0f / fmaxf(sqrtf(s), 1e-5f);
    }
    __syncthreads();

    // ---- Phase 4: x_new = invn * (attn @ v); accumulate out, write next xh ----
    {
        const int e0 = wave * 64;                       // 4 e-tiles per wave
        const _Float16* Vrow = vT + (size_t)(b * Dc + e0 + lo) * Sc; // +j*16*Sc

        v8f acc[4] = {};
#pragma unroll 4
        for (int kb = 0; kb < Sc; kb += 32) {
            v16h a = make_a(&attnh[lo][kb + hi * 16]);   // from LDS
            v16h bm[4];
#pragma unroll
            for (int j = 0; j < 4; j++)
                bm[j] = *(const v16h*)(Vrow + (size_t)j * 16 * Sc + kb + hi * 16);
#pragma unroll
            for (int j = 0; j < 4; j++)
                acc[j] = wmma_f16(a, bm[j], acc[j]);
        }

#pragma unroll
        for (int j = 0; j < 4; j++) {
            const int e = e0 + j * 16 + lo;
#pragma unroll
            for (int r = 0; r < 8; r++) {
                const int m = r + hi * 8;
                const float val = acc[j][r] * invn[m];  // fold L2-normalize here
                const size_t idx = (size_t)(b * Sc + q0 + m) * Dc + e;
                out[idx] += val;
                xh_next[idx] = (_Float16)val;
            }
        }
    }
}

// ---------------------------------------------------------------------------
// Inputs (setup_inputs order): 0=output f32 [B,S,D], 1=local_cor (unused),
// 2=event_type int [B,S], 3=W f32 [H,D,D], 4=b f32 [H,D]
// ---------------------------------------------------------------------------
extern "C" void kernel_launch(void* const* d_in, const int* in_sizes, int n_in,
                              void* d_out, int out_size, void* d_ws, size_t ws_size,
                              hipStream_t stream) {
    (void)in_sizes; (void)n_in; (void)out_size; (void)ws_size;
    const float* xin = (const float*)d_in[0];
    const int*   evt = (const int*)d_in[2];
    const float* Wf  = (const float*)d_in[3];
    const float* bf  = (const float*)d_in[4];
    float* out = (float*)d_out;

    char* ws = (char*)d_ws;
    const size_t nx = (size_t)Bc * Sc * Dc;              // 8,388,608 elements
    _Float16* xh0 = (_Float16*)(ws);                     // 16 MB
    _Float16* xh1 = (_Float16*)(ws + nx * 2);            // 16 MB
    _Float16* vT  = (_Float16*)(ws + 2 * nx * 2);        // 16 MB
    _Float16* Wh  = (_Float16*)(ws + 3 * nx * 2);        //  2 MB

    prep_kernel<<<(unsigned)((nx + 255) / 256), 256, 0, stream>>>(xin, Wf, out, xh0, Wh);

    _Float16* cur = xh0;
    _Float16* nxt = xh1;
    for (int h = 0; h < Hc; h++) {
        value_kernel<<<1024, 256, 0, stream>>>(cur, Wh + (size_t)h * Dc * Dc,
                                               bf + (size_t)h * Dc, vT);
        attn_kernel<<<1024, 256, 0, stream>>>(cur, vT, evt, out, nxt);
        _Float16* t = cur; cur = nxt; nxt = t;
    }
}
